// MIPTBlock_13529146982871
// MI455X (gfx1250) — compile-verified, matrix operand order checked by reading, not run on hardware
//
#include <hip/hip_runtime.h>
#include <stdint.h>

// ---------------------------------------------------------------------------
// MIPT block for MI455X (gfx1250, wave32):
//   WMMA bf16 GEMMs + GLOBAL_LOAD_ASYNC_TO_LDS (ASYNCcnt) for activations
//   + TDM tensor_load_to_lds (TENSORcnt) for weight tiles, double-buffered.
// ---------------------------------------------------------------------------

typedef __bf16 v16bf __attribute__((ext_vector_type(16)));
typedef float  v8f   __attribute__((ext_vector_type(8)));
typedef uint32_t u32x4 __attribute__((ext_vector_type(4)));
typedef uint32_t u32x8 __attribute__((ext_vector_type(8)));

union FragB16 { uint4 q[2]; v16bf v; };
union FragF32 { float f[8];  v8f   v; };

enum { B_ = 4, T_ = 4096, D_ = 512, M_ = 16384, DFF_ = 2048, KC_ = 16 };

// float -> bf16 (round to nearest even)
__device__ __forceinline__ uint16_t f2bf(float f) {
  union { float f; uint32_t u; } c; c.f = f;
  uint32_t u = c.u;
  uint32_t r = u + 0x7FFFu + ((u >> 16) & 1u);
  return (uint16_t)(r >> 16);
}

// async 16B global -> LDS copy (GLOBAL_LOAD_ASYNC_TO_LDS_B128, ASYNCcnt)
__device__ __forceinline__ void async_copy_b128(const void* gptr, void* lds_ptr) {
  uint32_t lds_off = (uint32_t)(size_t)lds_ptr;          // LDS aperture: addr[31:0]
  unsigned long long ga = (unsigned long long)(size_t)gptr;
  asm volatile("global_load_async_to_lds_b128 %0, %1, off"
               :: "v"(lds_off), "v"(ga) : "memory");
}

// TDM: 2D tile load Global->LDS. D# per ISA 8.3/8.4 (2-group form, VADDR2/3=NULL).
// Tile: tile1 rows x tile0 elems of 2-byte data, row stride = stride0 elems.
__device__ __forceinline__ void tdm_load_2d(uint64_t gaddr, uint32_t ldsaddr,
                                            uint32_t dim0, uint32_t dim1,
                                            uint32_t tile0, uint32_t tile1,
                                            uint32_t stride0) {
  u32x4 g0;
  g0[0] = 1u;                                           // count=1 (user D#)
  g0[1] = ldsaddr;                                      // lds_addr
  g0[2] = (uint32_t)gaddr;                              // global_addr[31:0]
  g0[3] = (uint32_t)((gaddr >> 32) & 0x1FFFFFFu)        // global_addr[56:32]
          | (2u << 30);                                 // type=2 ("image")
  u32x8 g1;
  g1[0] = (1u << 16);                                   // data_size=1 (2 bytes)
  g1[1] = (dim0 & 0xFFFFu) << 16;                       // tensor_dim0[15:0]
  g1[2] = ((dim0 >> 16) & 0xFFFFu) | ((dim1 & 0xFFFFu) << 16);
  g1[3] = ((dim1 >> 16) & 0xFFFFu) | (tile0 << 16);     // tile_dim0
  g1[4] = tile1 & 0xFFFFu;                              // tile_dim1 (tile_dim2=0)
  g1[5] = stride0;                                      // tensor_dim0_stride[31:0]
  g1[6] = 0u;
  g1[7] = 0u;
  asm volatile("tensor_load_to_lds %0, %1" :: "s"(g0), "s"(g1) : "memory");
}

// ---------------------------------------------------------------------------
// Weight conversion: f32 W[K,N] row-major -> bf16 TRANSPOSED dst[(rowOff+n)*ld + k]
// ---------------------------------------------------------------------------
__global__ __launch_bounds__(256) void conv_w_kernel(
    const float* __restrict__ W, uint16_t* __restrict__ dst,
    int Kdim, int Ncols, int rowOff, int ldDst) {
  size_t i = (size_t)blockIdx.x * 256 + threadIdx.x;
  if (i < (size_t)Kdim * Ncols) {
    int k = (int)(i / Ncols), n = (int)(i % Ncols);
    dst[(size_t)(rowOff + n) * ldDst + k] = f2bf(W[i]);
  }
}

__global__ __launch_bounds__(256) void copy_bias_kernel(
    const float* __restrict__ b, float* __restrict__ dst, int n, int off) {
  int i = blockIdx.x * 256 + threadIdx.x;
  if (i < n) dst[off + i] = b[i];
}

// ---------------------------------------------------------------------------
// LayerNorm (D=512) -> bf16 row; one 256-thread block per row
// ---------------------------------------------------------------------------
__global__ __launch_bounds__(256) void layernorm_to_bf16(
    const float* __restrict__ x, const float* __restrict__ g,
    const float* __restrict__ bt, uint16_t* __restrict__ out) {
  __shared__ float red[256];
  const int tid = threadIdx.x;
  const float* xr = x + (size_t)blockIdx.x * D_;
  float v0 = xr[tid], v1 = xr[tid + 256];
  red[tid] = v0 + v1;
  __syncthreads();
  for (int o = 128; o > 0; o >>= 1) {
    if (tid < o) red[tid] += red[tid + o];
    __syncthreads();
  }
  float mu = red[0] * (1.0f / D_);
  __syncthreads();
  float d0 = v0 - mu, d1 = v1 - mu;
  red[tid] = d0 * d0 + d1 * d1;
  __syncthreads();
  for (int o = 128; o > 0; o >>= 1) {
    if (tid < o) red[tid] += red[tid + o];
    __syncthreads();
  }
  float rstd = rsqrtf(red[0] * (1.0f / D_) + 1e-5f);
  uint16_t* orow = out + (size_t)blockIdx.x * D_;
  orow[tid]       = f2bf(d0 * rstd * g[tid]       + bt[tid]);
  orow[tid + 256] = f2bf(d1 * rstd * g[tid + 256] + bt[tid + 256]);
}

// ---------------------------------------------------------------------------
// WMMA bf16 GEMM: out[M,N] = epi(A[M,K] @ W^T[N,K] + bias) (+ resid)
// Block tile 128x64, 8 wave32s (4 M x 2 N), each wave 32x32 = 4 WMMAs.
// A tiles: async-to-LDS copies. B tiles: TDM (wave 0 only). Double buffered.
// ---------------------------------------------------------------------------
enum { EPI_NONE = 0, EPI_SIGMOID_PART = 1, EPI_GELU = 2 };

template <int EPI, bool OUT_BF16, bool RESID>
__global__ __launch_bounds__(256) void gemm_wmma_bf16(
    const uint16_t* __restrict__ A, int lda,
    const uint16_t* __restrict__ Wt, int N,   // Wt is [N][K] (pre-transposed)
    const float* __restrict__ bias, int K, int sigCols,
    float* __restrict__ outF, uint16_t* __restrict__ outB,
    const float* __restrict__ resid) {
  __shared__ __align__(16) uint16_t At[2][128 * 32];   // 2 x 8KB
  __shared__ __align__(16) uint16_t Bt[2][64 * 32];    // 2 x 4KB

  const int tid  = threadIdx.x;
  const int lane = tid & 31;
  const int wid  = tid >> 5;
  const int wm   = wid >> 1;        // 0..3 -> 32-row band
  const int wn   = wid & 1;         // 0..1 -> 32-col band
  const int mBase = blockIdx.y * 128;
  const int nBase = blockIdx.x * 64;
  const int l    = lane & 15;
  const int half = lane >> 4;

  // per-thread A-copy coordinates (2 x 16B per tile)
  const int cr = tid >> 2;          // 0..63
  const int ck = (tid & 3) * 8;     // 0,8,16,24
  const uint16_t* gA0 = A + (size_t)(mBase + cr) * lda + ck;
  const uint16_t* gA1 = A + (size_t)(mBase + 64 + cr) * lda + ck;
  const int lA0 = cr * 32 + ck;
  const int lA1 = (64 + cr) * 32 + ck;
  const uint64_t gBbase = (uint64_t)(size_t)(Wt + (size_t)nBase * K);

  FragF32 acc00, acc01, acc10, acc11;
#pragma unroll
  for (int i = 0; i < 8; i++) {
    acc00.f[i] = 0.f; acc01.f[i] = 0.f; acc10.f[i] = 0.f; acc11.f[i] = 0.f;
  }

  const int nT = K / 32;
  // prologue: tile 0 into buffer 0
  async_copy_b128(gA0, &At[0][lA0]);
  async_copy_b128(gA1, &At[0][lA1]);
  if (wid == 0)
    tdm_load_2d(gBbase, (uint32_t)(size_t)&Bt[0][0], 32u, 64u, 32u, 64u, (uint32_t)K);

  for (int i = 0; i < nT; i++) {
    const int cur = i & 1, nxt = cur ^ 1;
    const int k0 = i * 32;
    if (i + 1 < nT) {                    // pipeline next tile into other buffer
      async_copy_b128(gA0 + k0 + 32, &At[nxt][lA0]);
      async_copy_b128(gA1 + k0 + 32, &At[nxt][lA1]);
      if (wid == 0)
        tdm_load_2d(gBbase + (uint64_t)(k0 + 32) * 2u,
                    (uint32_t)(size_t)&Bt[nxt][0], 32u, 64u, 32u, 64u, (uint32_t)K);
      asm volatile("s_wait_asynccnt 0x2" ::: "memory");  // cur's 2 copies done
      if (wid == 0) __builtin_amdgcn_s_wait_tensorcnt(1);
    } else {
      asm volatile("s_wait_asynccnt 0x0" ::: "memory");
      if (wid == 0) __builtin_amdgcn_s_wait_tensorcnt(0);
    }
    __syncthreads();                     // cur tile visible to all waves

    // ISA 7.12.2 16-bit layouts: lanes 0-15 -> K {0..7,16..23}, 16-31 -> {8..15,24..31}
    FragB16 a0, a1, b0, b1;
    const uint16_t* ap0 = &At[cur][(wm * 32 + l) * 32 + half * 8];
    const uint16_t* ap1 = &At[cur][(wm * 32 + 16 + l) * 32 + half * 8];
    const uint16_t* bp0 = &Bt[cur][(wn * 32 + l) * 32 + half * 8];
    const uint16_t* bp1 = &Bt[cur][(wn * 32 + 16 + l) * 32 + half * 8];
    a0.q[0] = *(const uint4*)ap0; a0.q[1] = *(const uint4*)(ap0 + 16);
    a1.q[0] = *(const uint4*)ap1; a1.q[1] = *(const uint4*)(ap1 + 16);
    b0.q[0] = *(const uint4*)bp0; b0.q[1] = *(const uint4*)(bp0 + 16);
    b1.q[0] = *(const uint4*)bp1; b1.q[1] = *(const uint4*)(bp1 + 16);

    acc00.v = __builtin_amdgcn_wmma_f32_16x16x32_bf16(false, a0.v, false, b0.v,
                                                      (short)0, acc00.v, false, false);
    acc01.v = __builtin_amdgcn_wmma_f32_16x16x32_bf16(false, a0.v, false, b1.v,
                                                      (short)0, acc01.v, false, false);
    acc10.v = __builtin_amdgcn_wmma_f32_16x16x32_bf16(false, a1.v, false, b0.v,
                                                      (short)0, acc10.v, false, false);
    acc11.v = __builtin_amdgcn_wmma_f32_16x16x32_bf16(false, a1.v, false, b1.v,
                                                      (short)0, acc11.v, false, false);
    __syncthreads();                     // all frag reads done before overwrite
  }

  // C/D layout: VGPR v -> row = base + (lane<16?0:8) + v, col = base + (lane&15)
#pragma unroll
  for (int sm = 0; sm < 2; sm++) {
#pragma unroll
    for (int sn = 0; sn < 2; sn++) {
      FragF32& acc = sm ? (sn ? acc11 : acc10) : (sn ? acc01 : acc00);
      const int row0 = mBase + wm * 32 + sm * 16 + half * 8;
      const int col  = nBase + wn * 32 + sn * 16 + l;
      const float bv = bias[col];
#pragma unroll
      for (int v = 0; v < 8; v++) {
        int row = row0 + v;
        float xv = acc.f[v] + bv;
        if (EPI == EPI_SIGMOID_PART) {
          if (col < sigCols) xv = 1.f / (1.f + __expf(-xv));
        }
        if (EPI == EPI_GELU) xv = 0.5f * xv * (1.f + erff(xv * 0.70710678118654752f));
        size_t o = (size_t)row * N + col;
        if (RESID) xv += resid[o];
        if (OUT_BF16) outB[o] = f2bf(xv); else outF[o] = xv;
      }
    }
  }
}

// ---------------------------------------------------------------------------
// MIPT scan: 13 shift-by-1 combine steps, fully fused using a halo-13 tile.
// proj layout per row (2048): [p | theta | inp_re | inp_im] (p already sigmoid)
// ---------------------------------------------------------------------------
#define SCAN_T 128
#define SCAN_D 16
#define SCAN_H 13
#define SCAN_L (SCAN_T + SCAN_H)   // 141
#define SCAN_E (SCAN_L * SCAN_D)   // 2256

__global__ __launch_bounds__(256) void mipt_scan_kernel(
    const float* __restrict__ proj, float* __restrict__ h_re) {
  __shared__ float sAr[SCAN_E], sAi[SCAN_E], sBr[SCAN_E], sBi[SCAN_E];
  const int d0 = blockIdx.x * SCAN_D;
  const int t0 = blockIdx.y * SCAN_T;
  const int bb = blockIdx.z;
  const int tid = threadIdx.x;

  for (int e = tid; e < SCAN_E; e += 256) {
    int i = e >> 4, dl = e & 15;
    int t = t0 - SCAN_H + i;
    float ar = 0.f, ai = 0.f, br = 0.f, bi = 0.f;
    if (t >= 0) {
      size_t rb = ((size_t)bb * T_ + t) * (size_t)DFF_ + d0 + dl;
      float p  = proj[rb];
      float th = proj[rb + 512];
      br = proj[rb + 1024];
      bi = proj[rb + 1536];
      float om = 1.f - p;
      float sn, cs;
      __sincosf(th, &sn, &cs);
      ar = om * cs; ai = om * sn;
    }
    sAr[e] = ar; sAi[e] = ai; sBr[e] = br; sBi[e] = bi;
  }
  __syncthreads();

  float nar[9], nai[9], nbr[9], nbi[9];
  bool upd[9];
  for (int step = 0; step < 13; step++) {   // ceil(log2(4097)) = 13
#pragma unroll
    for (int j = 0; j < 9; j++) {
      int e = tid + j * 256;
      upd[j] = false;
      if (e < SCAN_E) {
        int i = e >> 4;
        int t = t0 - SCAN_H + i;
        if (i >= 1 && t >= 1) {             // mask: t >= 1; i>=1 has in-tile pred
          float ar = sAr[e], ai = sAi[e], br = sBr[e], bi = sBi[e];
          float arp = sAr[e - SCAN_D], aip = sAi[e - SCAN_D];
          float brp = sBr[e - SCAN_D], bip = sBi[e - SCAN_D];
          nar[j] = ar * arp - ai * aip;
          nai[j] = ar * aip + ai * arp;
          nbr[j] = ar * brp - ai * bip + br;
          nbi[j] = ar * bip + ai * brp + bi;
          upd[j] = true;
        }
      }
    }
    __syncthreads();
#pragma unroll
    for (int j = 0; j < 9; j++) {
      if (upd[j]) {
        int e = tid + j * 256;
        sAr[e] = nar[j]; sAi[e] = nai[j]; sBr[e] = nbr[j]; sBi[e] = nbi[j];
      }
    }
    __syncthreads();
  }

  for (int e = tid; e < SCAN_E; e += 256) {
    int i = e >> 4;
    if (i >= SCAN_H) {                       // interior is exact after 13 steps
      int t = t0 + (i - SCAN_H);
      h_re[((size_t)bb * T_ + t) * D_ + d0 + (e & 15)] = sBr[e];
    }
  }
}

// ---------------------------------------------------------------------------
// p_scalar[b,t] = mean over D of p (cols 0..511 of proj row)
// ---------------------------------------------------------------------------
__global__ __launch_bounds__(256) void pscalar_kernel(
    const float* __restrict__ proj, float* __restrict__ ps) {
  __shared__ float red[256];
  const int tid = threadIdx.x;
  const float* pr = proj + (size_t)blockIdx.x * DFF_;
  red[tid] = pr[tid] + pr[tid + 256];
  __syncthreads();
  for (int o = 128; o > 0; o >>= 1) {
    if (tid < o) red[tid] += red[tid + o];
    __syncthreads();
  }
  if (tid == 0) ps[blockIdx.x] = red[0] * (1.0f / D_);
}

// h_mean[b,d] = mean over T of h_re ; grid = B*2 blocks, coalesced over d
__global__ __launch_bounds__(256) void hmean_kernel(
    const float* __restrict__ h_re, float* __restrict__ hm) {
  int b = blockIdx.x >> 1;
  int d = (blockIdx.x & 1) * 256 + threadIdx.x;
  float s = 0.f;
  for (int t = 0; t < T_; t++) s += h_re[((size_t)b * T_ + t) * D_ + d];
  hm[(size_t)b * D_ + d] = s * (1.0f / T_);
}

// h = h_re + x (residual)
__global__ __launch_bounds__(256) void add_kernel(
    const float* __restrict__ a, const float* __restrict__ b,
    float* __restrict__ o, size_t n) {
  size_t i = (size_t)blockIdx.x * 256 + threadIdx.x;
  if (i < n) o[i] = a[i] + b[i];
}

// ---------------------------------------------------------------------------
// Top-K (K=16) per batch over T=4096 (iterative argmax, lowest-index tiebreak)
// ---------------------------------------------------------------------------
__global__ __launch_bounds__(256) void topk_kernel(
    const float* __restrict__ ps, int* __restrict__ idxOut) {
  __shared__ float vals[T_];
  __shared__ float rv[256];
  __shared__ int   ri[256];
  const int b = blockIdx.x, tid = threadIdx.x;
  for (int t = tid; t < T_; t += 256) vals[t] = ps[(size_t)b * T_ + t];
  __syncthreads();
  for (int k = 0; k < KC_; k++) {
    float bv = -3.4e38f; int bi = 0;
    for (int t = tid; t < T_; t += 256) {
      float v = vals[t];
      if (v > bv) { bv = v; bi = t; }
    }
    rv[tid] = bv; ri[tid] = bi;
    __syncthreads();
    for (int o = 128; o > 0; o >>= 1) {
      if (tid < o) {
        float v2 = rv[tid + o]; int i2 = ri[tid + o];
        if (v2 > rv[tid] || (v2 == rv[tid] && i2 < ri[tid])) { rv[tid] = v2; ri[tid] = i2; }
      }
      __syncthreads();
    }
    if (tid == 0) { idxOut[b * KC_ + k] = ri[0]; vals[ri[0]] = -3.4e38f; }
    __syncthreads();
  }
}

// keys/values = h_topk @ Wk/Wv + bias ; one block per (b,k) row
__global__ __launch_bounds__(256) void cache_kv_kernel(
    const float* __restrict__ h_re, const int* __restrict__ topk,
    const float* __restrict__ Wk, const float* __restrict__ kb,
    const float* __restrict__ Wv, const float* __restrict__ vb,
    float* __restrict__ keys, float* __restrict__ values) {
  __shared__ float row[D_];
  const int b = blockIdx.x / KC_, k = blockIdx.x % KC_;
  const int t = topk[b * KC_ + k];
  const float* hr = h_re + ((size_t)b * T_ + t) * D_;
  row[threadIdx.x] = hr[threadIdx.x];
  row[threadIdx.x + 256] = hr[threadIdx.x + 256];
  __syncthreads();
  for (int c = threadIdx.x; c < D_; c += 256) {
    float sk = kb[c], sv = vb[c];
    for (int j = 0; j < D_; j++) {
      float h = row[j];
      sk += h * Wk[(size_t)j * D_ + c];
      sv += h * Wv[(size_t)j * D_ + c];
    }
    keys[((size_t)b * KC_ + k) * D_ + c] = sk;
    values[((size_t)b * KC_ + k) * D_ + c] = sv;
  }
}

// query = h_mean @ Wq + bias ; one block per batch
__global__ __launch_bounds__(256) void query_kernel(
    const float* __restrict__ hm, const float* __restrict__ Wq,
    const float* __restrict__ qb, float* __restrict__ query) {
  __shared__ float row[D_];
  const int b = blockIdx.x;
  row[threadIdx.x] = hm[(size_t)b * D_ + threadIdx.x];
  row[threadIdx.x + 256] = hm[(size_t)b * D_ + threadIdx.x + 256];
  __syncthreads();
  for (int c = threadIdx.x; c < D_; c += 256) {
    float s = qb[c];
    for (int j = 0; j < D_; j++) s += row[j] * Wq[(size_t)j * D_ + c];
    query[(size_t)b * D_ + c] = s;
  }
}

// attention over 16 cached slots + gate + combine -> out[b,d]
__global__ __launch_bounds__(256) void attn_out_kernel(
    const float* __restrict__ hm, const float* __restrict__ query,
    const float* __restrict__ keys, const float* __restrict__ values,
    const float* __restrict__ gw, const float* __restrict__ gb,
    float* __restrict__ outv) {
  __shared__ float q[D_];
  __shared__ float sc[KC_];
  __shared__ float red[256];
  const int b = blockIdx.x, tid = threadIdx.x;
  q[tid] = query[(size_t)b * D_ + tid];
  q[tid + 256] = query[(size_t)b * D_ + tid + 256];
  __syncthreads();
  if (tid < KC_) {
    const float* kr = keys + ((size_t)b * KC_ + tid) * D_;
    float s = 0.f;
    for (int j = 0; j < D_; j++) s += q[j] * kr[j];
    sc[tid] = s * 0.044194173824159216f;  // 1/sqrt(512)
  }
  red[tid] = hm[(size_t)b * D_ + tid] * gw[tid] +
             hm[(size_t)b * D_ + tid + 256] * gw[tid + 256];
  __syncthreads();
  for (int o = 128; o > 0; o >>= 1) {
    if (tid < o) red[tid] += red[tid + o];
    __syncthreads();
  }
  float g = 1.f / (1.f + __expf(-(red[0] + gb[0])));
  if (tid == 0) {
    float m = sc[0];
    for (int k = 1; k < KC_; k++) m = fmaxf(m, sc[k]);
    float ssum = 0.f;
    for (int k = 0; k < KC_; k++) { sc[k] = __expf(sc[k] - m); ssum += sc[k]; }
    float inv = 1.f / ssum;
    for (int k = 0; k < KC_; k++) sc[k] *= inv;
  }
  __syncthreads();
  for (int d = tid; d < D_; d += 256) {
    float c = 0.f;
    for (int k = 0; k < KC_; k++) c += sc[k] * values[((size_t)b * KC_ + k) * D_ + d];
    outv[(size_t)b * D_ + d] = hm[(size_t)b * D_ + d] + g * c;
  }
}

// ---------------------------------------------------------------------------
// Workspace layout (all offsets 256B aligned)
// ---------------------------------------------------------------------------
static const size_t OFF_XN     = 0;                         // bf16 [16384,512]
static const size_t OFF_WFUSED = 16777216;                  // bf16 [2048,512] (N,K)
static const size_t OFF_WF1    = OFF_WFUSED + 2097152;      // bf16 [2048,512] (N,K)
static const size_t OFF_WF2    = OFF_WF1    + 2097152;      // bf16 [512,2048] (N,K)
static const size_t OFF_BFUSED = OFF_WF2    + 2097152;      // f32 [2048]
static const size_t OFF_PROJ   = OFF_BFUSED + 8192;         // f32 [16384,2048]
static const size_t OFF_HRE    = OFF_PROJ   + 134217728;    // f32 [16384,512]
static const size_t OFF_FFN1A  = OFF_HRE    + 33554432;     // bf16 [16384,2048]
static const size_t OFF_PS     = OFF_FFN1A  + 67108864;     // f32 [4,4096]
static const size_t OFF_HM     = OFF_PS     + 65536;        // f32 [4,512]
static const size_t OFF_TOPK   = OFF_HM     + 8192;         // int [4,16]
static const size_t OFF_KEYS   = OFF_TOPK   + 256;          // f32 [4,16,512]
static const size_t OFF_VALS   = OFF_KEYS   + 131072;       // f32 [4,16,512]
static const size_t OFF_QUERY  = OFF_VALS   + 131072;       // f32 [4,512]

extern "C" void kernel_launch(void* const* d_in, const int* in_sizes, int n_in,
                              void* d_out, int out_size, void* d_ws, size_t ws_size,
                              hipStream_t stream) {
  (void)in_sizes; (void)n_in; (void)out_size; (void)ws_size;
  const float* x     = (const float*)d_in[0];
  const float* ln1_g = (const float*)d_in[1];
  const float* ln1_b = (const float*)d_in[2];
  const float* ln2_g = (const float*)d_in[3];
  const float* ln2_b = (const float*)d_in[4];
  const float* Wp_w  = (const float*)d_in[5];  const float* Wp_b = (const float*)d_in[6];
  const float* Wt_w  = (const float*)d_in[7];  const float* Wt_b = (const float*)d_in[8];
  const float* Wr_w  = (const float*)d_in[9];  const float* Wr_b = (const float*)d_in[10];
  const float* Wi_w  = (const float*)d_in[11]; const float* Wi_b = (const float*)d_in[12];
  const float* Wk_w  = (const float*)d_in[13]; const float* Wk_b = (const float*)d_in[14];
  const float* Wv_w  = (const float*)d_in[15]; const float* Wv_b = (const float*)d_in[16];
  const float* Wq_w  = (const float*)d_in[17]; const float* Wq_b = (const float*)d_in[18];
  const float* gate_w = (const float*)d_in[19]; const float* gate_b = (const float*)d_in[20];
  const float* f1_w  = (const float*)d_in[21]; const float* f1_b = (const float*)d_in[22];
  const float* f2_w  = (const float*)d_in[23]; const float* f2_b = (const float*)d_in[24];

  char* ws = (char*)d_ws;
  uint16_t* xn      = (uint16_t*)(ws + OFF_XN);
  uint16_t* wfused  = (uint16_t*)(ws + OFF_WFUSED);
  uint16_t* wf1     = (uint16_t*)(ws + OFF_WF1);
  uint16_t* wf2     = (uint16_t*)(ws + OFF_WF2);
  float*    bfused  = (float*)(ws + OFF_BFUSED);
  float*    proj    = (float*)(ws + OFF_PROJ);
  float*    h_re    = (float*)(ws + OFF_HRE);
  uint16_t* ffn1a   = (uint16_t*)(ws + OFF_FFN1A);
  float*    ps      = (float*)(ws + OFF_PS);
  float*    hm      = (float*)(ws + OFF_HM);
  int*      topk    = (int*)(ws + OFF_TOPK);
  float*    keys    = (float*)(ws + OFF_KEYS);
  float*    vals    = (float*)(ws + OFF_VALS);
  float*    query   = (float*)(ws + OFF_QUERY);

  float* h_out   = (float*)d_out;                       // [B,T,D]
  float* out_vec = h_out + (size_t)B_ * T_ * D_;        // [B,D]

  // --- weight/bias conversion (bf16, pre-transposed to [N][K]) ---
  conv_w_kernel<<<1024, 256, 0, stream>>>(Wp_w, wfused, D_, D_, 0,    D_);
  conv_w_kernel<<<1024, 256, 0, stream>>>(Wt_w, wfused, D_, D_, 512,  D_);
  conv_w_kernel<<<1024, 256, 0, stream>>>(Wr_w, wfused, D_, D_, 1024, D_);
  conv_w_kernel<<<1024, 256, 0, stream>>>(Wi_w, wfused, D_, D_, 1536, D_);
  conv_w_kernel<<<4096, 256, 0, stream>>>(f1_w, wf1, D_, DFF_, 0, D_);
  conv_w_kernel<<<4096, 256, 0, stream>>>(f2_w, wf2, DFF_, D_, 0, DFF_);
  copy_bias_kernel<<<2, 256, 0, stream>>>(Wp_b, bfused, D_, 0);
  copy_bias_kernel<<<2, 256, 0, stream>>>(Wt_b, bfused, D_, 512);
  copy_bias_kernel<<<2, 256, 0, stream>>>(Wr_b, bfused, D_, 1024);
  copy_bias_kernel<<<2, 256, 0, stream>>>(Wi_b, bfused, D_, 1536);

  // --- LN1 -> xn (bf16) ---
  layernorm_to_bf16<<<M_, 256, 0, stream>>>(x, ln1_g, ln1_b, xn);

  // --- fused projection GEMM: [16384,512]@[512,2048], sigmoid on p cols ---
  gemm_wmma_bf16<EPI_SIGMOID_PART, false, false>
      <<<dim3(DFF_ / 64, M_ / 128), 256, 0, stream>>>(
          xn, D_, wfused, DFF_, bfused, D_, 512, proj, nullptr, nullptr);

  // --- fused 13-step scan -> h_re ---
  mipt_scan_kernel<<<dim3(D_ / SCAN_D, T_ / SCAN_T, B_), 256, 0, stream>>>(proj, h_re);

  // --- reductions + residual ---
  pscalar_kernel<<<M_, 256, 0, stream>>>(proj, ps);
  hmean_kernel<<<B_ * 2, 256, 0, stream>>>(h_re, hm);
  add_kernel<<<(unsigned)(((size_t)M_ * D_ + 255) / 256), 256, 0, stream>>>(
      h_re, x, h_out, (size_t)M_ * D_);

  // --- sparse cache attention path ---
  topk_kernel<<<B_, 256, 0, stream>>>(ps, topk);
  cache_kv_kernel<<<B_ * KC_, 256, 0, stream>>>(h_re, topk, Wk_w, Wk_b, Wv_w, Wv_b, keys, vals);
  query_kernel<<<B_, 256, 0, stream>>>(hm, Wq_w, Wq_b, query);
  attn_out_kernel<<<B_, 256, 0, stream>>>(hm, query, keys, vals, gate_w, gate_b, out_vec);

  // --- FFN: LN2 -> gelu(hn@W1+b1)@W2+b2 + h ---
  layernorm_to_bf16<<<M_, 256, 0, stream>>>(h_out, ln2_g, ln2_b, xn);  // reuse xn buf
  gemm_wmma_bf16<EPI_GELU, true, false>
      <<<dim3(DFF_ / 64, M_ / 128), 256, 0, stream>>>(
          xn, D_, wf1, DFF_, f1_b, D_, 0, nullptr, ffn1a, nullptr);
  gemm_wmma_bf16<EPI_NONE, false, true>
      <<<dim3(D_ / 64, M_ / 128), 256, 0, stream>>>(
          ffn1a, DFF_, wf2, D_, f2_b, DFF_, 0, h_out, nullptr, h_out);
}
